// ContinuousFilterConv_47974784696382
// MI455X (gfx1250) — compile-verified, matrix operand order checked by reading, not run on hardware
//
#include <hip/hip_runtime.h>
#include <hip/hip_bf16.h>

#define N_NODES 20000
#define N_EDGES 8192
#define UNITS   128
#define NUM_G   50
#define CUTOFF  8.0f
#define GAMMA   10.0f

// GEMM K dimension: 128*128 (h (x) t) + 128 (bias * t) = 16512 -> 516 chunks of 32
#define KCHUNKS 516

typedef __attribute__((ext_vector_type(16))) __bf16       v16bf;
typedef __attribute__((ext_vector_type(8)))  float        v8f;
typedef __attribute__((ext_vector_type(8)))  unsigned int v8u;

__global__ __launch_bounds__(256) void zero_kernel(float* p, int n) {
  int i = blockIdx.x * 256 + threadIdx.x;
  if (i < n) p[i] = 0.0f;
}

// Pack W2 (+ b2 as 128 extra K-rows) into WMMA-B fragment-major bf16 layout.
// Block b = kc*8 + ntile holds the 32x16 B-tile (K = kc*32.., N = ntile*16..)
// as per-lane-contiguous halves: BB[b*512 + lane*16 + r], where
// lane -> N = ntile*16 + (lane&15), K = kc*32 + (lane&16) + r.
// Contiguous k order (not AMD's table order), matched by the A operand of
// v_wmma_f32_16x16x32_bf16, so each lane's fragment is two coalesced
// global_load_b128.
__global__ __launch_bounds__(256) void pack_b_kernel(const float* __restrict__ W2,
                                                     const float* __restrict__ b2,
                                                     __bf16* __restrict__ BB) {
  int tid = blockIdx.x * 256 + threadIdx.x;
  const int total = KCHUNKS * 8 * 512;
  if (tid >= total) return;
  int b    = tid >> 9;
  int lane = (tid >> 4) & 31;
  int r    = tid & 15;
  int nt   = b & 7;
  int kc   = b >> 3;
  int kk   = kc * 32 + (lane & 16) + r;   // global K index in [0, 16512)
  int i    = nt * 16 + (lane & 15);       // output column in [0, 128)
  float v;
  if (kk < UNITS * UNITS) {
    int k = kk >> 7, j = kk & 127;
    v = W2[(size_t)k * (UNITS * UNITS) + i * UNITS + j];
  } else {
    int j = kk - UNITS * UNITS;
    v = b2[i * UNITS + j];
  }
  BB[tid] = (__bf16)v;
}

// Per-edge preprocessing: gaussian basis -> h = silu(df@W1+b1),
// t = (nf[src] @ Wt) * mask.  Stored bf16 for the WMMA GEMM.
__global__ __launch_bounds__(128) void edge_pre_kernel(
    const float* __restrict__ nf, const int* __restrict__ ei,
    const float* __restrict__ dist,
    const float* __restrict__ W1, const float* __restrict__ b1,
    const float* __restrict__ Wt,
    __bf16* __restrict__ hout, __bf16* __restrict__ tout) {
  int e = blockIdx.x;
  int u = threadIdx.x;
  __shared__ float df_s[NUM_G];
  __shared__ float src_s[UNITS];
  float d = dist[e];
  if (u < NUM_G) {
    float c = (float)u * (30.0f / 49.0f);   // linspace(0, 30, 50)
    float x = d - c;
    df_s[u] = __expf(-GAMMA * x * x);
  }
  int s = ei[e];                            // edge_indices[0][e]
  src_s[u] = nf[(size_t)s * UNITS + u];
  __syncthreads();
  float a1 = b1[u];
#pragma unroll 5
  for (int g = 0; g < NUM_G; ++g) a1 += df_s[g] * W1[g * UNITS + u];
  float h = a1 / (1.0f + __expf(-a1));      // silu
  float a2 = 0.0f;
#pragma unroll 8
  for (int v = 0; v < UNITS; ++v) a2 += src_s[v] * Wt[v * UNITS + u];
  float mask = (d <= CUTOFF) ? 1.0f : 0.0f;
  hout[(size_t)e * UNITS + u] = (__bf16)h;
  tout[(size_t)e * UNITS + u] = (__bf16)(a2 * mask);
}

// Fused GEMM: filtered[e,i] = sum_{k,j} h[e,k] t[e,j] W2r[k,i,j] + b2-term,
// as (on-the-fly outer-product A) x (pre-packed B) with
// v_wmma_f32_16x16x32_bf16, then scatter-add to destination nodes.
//
// 32 edges per workgroup (two M-tiles), 4 waves:
//   wave w -> M-tile (w>>1), N-tiles (w&1)*4 .. +3.
// Each A-fragment build (8x v_pk_mul_bf16 + 2 LDS b128 loads) now feeds
// FOUR WMMAs, and each B fragment is shared by two waves via WGP$/L2.
__global__ __launch_bounds__(128) void cfconv_gemm_kernel(
    const __bf16* __restrict__ hbuf, const __bf16* __restrict__ tbuf,
    const __bf16* __restrict__ BB, const int* __restrict__ ei,
    float* __restrict__ accbuf) {
  __shared__ unsigned int   t_lds[32 * 64];    // 32 edges x 128 bf16 (dword pairs)
  __shared__ unsigned short h_lds[32 * 128];   // 32 edges x 128 bf16
  __shared__ int            dst_lds[32];

  int tid = threadIdx.x;
  int e0  = blockIdx.x * 32;

  const unsigned int* tg = (const unsigned int*)(tbuf + (size_t)e0 * UNITS);
  for (int idx = tid; idx < 2048; idx += 128) t_lds[idx] = tg[idx];
  unsigned int* h32 = (unsigned int*)h_lds;
  const unsigned int* hg = (const unsigned int*)(hbuf + (size_t)e0 * UNITS);
  for (int idx = tid; idx < 2048; idx += 128) h32[idx] = hg[idx];
  if (tid < 32) dst_lds[tid] = ei[N_EDGES + e0 + tid];   // edge_indices[1][...]
  __syncthreads();

  int lane = tid & 31;
  int wave = tid >> 5;
  int mt   = wave >> 1;                 // M-tile (0/1) within the 32-edge block
  int ng   = (wave & 1) * 4;            // first of this wave's 4 N-tiles
  int eR   = mt * 16 + (lane & 15);     // local edge row for A / h reads
  int khalf = (lane & 16) ? 8 : 0;      // dword offset within a 32-wide K chunk

  v8f acc0 = {}, acc1 = {}, acc2 = {}, acc3 = {};
  const v8u* bbv = (const v8u*)BB;      // 32B per lane-fragment chunk

  for (int k = 0; k <= UNITS; ++k) {    // k==UNITS is the bias row (scale = 1.0)
    unsigned int s2 = (k < UNITS)
        ? ((unsigned int)h_lds[eR * UNITS + k] * 0x10001u)  // splat bf16 pair
        : 0x3F803F80u;                                      // bf16(1.0) pair
#pragma unroll
    for (int jc = 0; jc < 4; ++jc) {
      int kc = k * 4 + jc;              // global 32-wide K chunk index
      // A fragment: h[e,k] * t[e, jc*32 + (lane&16) + 0..15]  (packed bf16 mul)
      int tb = eR * 64 + jc * 16 + khalf;
      v8u au;
#pragma unroll
      for (int d2 = 0; d2 < 8; ++d2) {
        unsigned int td = t_lds[tb + d2];
        unsigned int rr;
        asm("v_pk_mul_bf16 %0, %1, %2" : "=v"(rr) : "v"(td), "v"(s2));
        au[d2] = rr;
      }
      v16bf a = __builtin_bit_cast(v16bf, au);

      int blk = (kc * 8 + ng) * 32 + lane;
      v8u b0u = bbv[blk];
      v8u b1u = bbv[blk + 32];
      v8u b2u = bbv[blk + 64];
      v8u b3u = bbv[blk + 96];
      if (kc + 1 < KCHUNKS)   // pull next K-chunk of B into the WGP cache
        __builtin_prefetch((const void*)&bbv[blk + 8 * 32], 0, 3);
      acc0 = __builtin_amdgcn_wmma_f32_16x16x32_bf16(
          false, a, false, __builtin_bit_cast(v16bf, b0u), (short)0, acc0, false, false);
      acc1 = __builtin_amdgcn_wmma_f32_16x16x32_bf16(
          false, a, false, __builtin_bit_cast(v16bf, b1u), (short)0, acc1, false, false);
      acc2 = __builtin_amdgcn_wmma_f32_16x16x32_bf16(
          false, a, false, __builtin_bit_cast(v16bf, b2u), (short)0, acc2, false, false);
      acc3 = __builtin_amdgcn_wmma_f32_16x16x32_bf16(
          false, a, false, __builtin_bit_cast(v16bf, b3u), (short)0, acc3, false, false);
    }
  }

  // Epilogue: C/D layout -> lane holds column (lane&15), rows r + (lane<16?0:8).
  int col0   = ng * 16 + (lane & 15);
  int rowoff = (lane & 16) ? 8 : 0;
#pragma unroll
  for (int r = 0; r < 8; ++r) {
    int el  = mt * 16 + r + rowoff;
    int dst = dst_lds[el];
    float* row = &accbuf[(size_t)dst * UNITS + col0];
    atomicAdd(row,      acc0[r]);
    atomicAdd(row + 16, acc1[r]);
    atomicAdd(row + 32, acc2[r]);
    atomicAdd(row + 48, acc3[r]);
  }
}

__global__ __launch_bounds__(256) void silu_kernel(const float* __restrict__ acc,
                                                   float* __restrict__ out, int n) {
  int i = blockIdx.x * 256 + threadIdx.x;
  if (i < n) {
    float x = acc[i];
    out[i] = x / (1.0f + __expf(-x));
  }
}

extern "C" void kernel_launch(void* const* d_in, const int* in_sizes, int n_in,
                              void* d_out, int out_size, void* d_ws, size_t ws_size,
                              hipStream_t stream) {
  const float* nf   = (const float*)d_in[0];
  const int*   ei   = (const int*)  d_in[1];
  const float* dist = (const float*)d_in[2];
  const float* W1   = (const float*)d_in[3];
  const float* b1   = (const float*)d_in[4];
  const float* W2   = (const float*)d_in[5];
  const float* b2   = (const float*)d_in[6];
  const float* Wt   = (const float*)d_in[7];
  float* out = (float*)d_out;

  // Workspace layout (bytes):
  //   BB   : KCHUNKS*8*512 bf16 = 4,227,072
  //   hbuf : E*U bf16           = 2,097,152
  //   tbuf : E*U bf16           = 2,097,152
  //   acc  : N_NODES*U f32      = 10,240,000   (total ~18.7 MB)
  char*   ws   = (char*)d_ws;
  __bf16* BB   = (__bf16*)(ws);
  __bf16* hbuf = (__bf16*)(ws + 4227072);
  __bf16* tbuf = (__bf16*)(ws + 4227072 + 2097152);
  float*  acc  = (float*) (ws + 4227072 + 2 * 2097152);

  const int nacc = N_NODES * UNITS;
  zero_kernel<<<(nacc + 255) / 256, 256, 0, stream>>>(acc, nacc);
  pack_b_kernel<<<(KCHUNKS * 8 * 512 + 255) / 256, 256, 0, stream>>>(W2, b2, BB);
  edge_pre_kernel<<<N_EDGES, 128, 0, stream>>>(nf, ei, dist, W1, b1, Wt, hbuf, tbuf);
  cfconv_gemm_kernel<<<N_EDGES / 32, 128, 0, stream>>>(hbuf, tbuf, BB, ei, acc);
  silu_kernel<<<(nacc + 255) / 256, 256, 0, stream>>>(acc, out, nacc);
}
